// SparseAttention_48387101557379
// MI455X (gfx1250) — matrix operand order repaired and checked by reading, not compiled, not run on hardware
//
#include <hip/hip_runtime.h>
#include <hip/hip_bf16.h>

#define D_MODEL   512
#define NUM_HEADS 8
#define HEAD_DIM  64
#define WINDOW    64
#define BATCH     2
#define SEQ       4096
#define M_TOTAL   (BATCH * SEQ)   // 8192
#define KT_SPAN   9               // key tiles covering the band per 16-query tile
#define PCOLS     160             // 9*16 = 144 prob columns, padded to 5*32

typedef __bf16 bf16_t;
typedef unsigned int u32;
typedef __attribute__((ext_vector_type(16))) __bf16 v16bf;
typedef __attribute__((ext_vector_type(8)))  __bf16 v8bf;
typedef __attribute__((ext_vector_type(4)))  __bf16 v4bf;
typedef __attribute__((ext_vector_type(8)))  float  v8f;
typedef __attribute__((ext_vector_type(4)))  float  v4f;
typedef __attribute__((ext_vector_type(4)))  u32    u32x4;
typedef __attribute__((ext_vector_type(8)))  int    i32x8;
typedef __attribute__((ext_vector_type(4)))  int    i32x4;

#define LDS_OFF(p) ((u32)(size_t)(p))   // low 32 bits of generic LDS pointer = LDS byte offset

static __device__ __forceinline__ v16bf cat8(v8bf lo, v8bf hi) {
  return __builtin_shufflevector(lo, hi, 0,1,2,3,4,5,6,7,8,9,10,11,12,13,14,15);
}

// A-fragment loader for v_wmma_f32_16x16x32_bf16 (16-bit A 16x32, contiguous k order, not the ISA table order; B matches):
//   lane<16 : K = {k0..k0+15};  lane>=16: +16
// `row` must already point at this lane's row (M = lane&15).
static __device__ __forceinline__ v16bf load_a_frag(const bf16_t* row, int k0, int half) {
  v8bf lo = *(const v8bf*)(row + k0 + half * 16);
  v8bf hi = *(const v8bf*)(row + k0 + half * 16 + 8);
  return cat8(lo, hi);
}

// ---------------------------------------------------------------------------
// TDM: DMA a 64-row x 32-col bf16 tile (row stride 512 elements) from global
// memory into LDS, packed row-major (64x32). Tracked with TENSORcnt.
// D# layout per CDNA5 ISA.
// ---------------------------------------------------------------------------
static __device__ __forceinline__ void tdm_load_tile_64x32(const bf16_t* gsrc, u32 lds_off) {
  unsigned long long ga = (unsigned long long)(size_t)gsrc;
  u32x4 g0;
  g0[0] = 1u;                               // count=1, user descriptor
  g0[1] = lds_off;                          // lds_addr (bytes)
  g0[2] = (u32)ga;                          // global_addr[31:0]
  g0[3] = (u32)(ga >> 32) | (2u << 30);     // global_addr[56:32] | type=2 ("image")
  i32x8 g1;
  g1[0] = (1 << 16);                        // wg_mask=0 | data_size=1 (2B) | no flags
  g1[1] = (int)(512u << 16);                // atomic_barrier=0 | tensor_dim0[15:0]=512
  g1[2] = (int)(512u << 16);                // tensor_dim0[31:16]=0 | tensor_dim1[15:0]=512
  g1[3] = (int)(32u << 16);                 // tensor_dim1[31:16]=0 | tile_dim0=32
  g1[4] = 64;                               // tile_dim1=64 | tile_dim2=0
  g1[5] = 512;                              // tensor_dim0_stride[31:0]=512 (elements)
  g1[6] = 0;                                // dim0_stride hi | dim1_stride lo (unused, 2D)
  g1[7] = 0;
  i32x4 gz = {0, 0, 0, 0};                  // groups 2/3 unused (2D tile)
#if __has_include(<hip/amd_detail/amd_gfx1250_TDM.h>)
  i32x8 gz8 = {0, 0, 0, 0, 0, 0, 0, 0};
  __builtin_amdgcn_tensor_load_to_lds(g0, g1, gz, gz, gz8, 0);
#else
  __builtin_amdgcn_tensor_load_to_lds(g0, g1, gz, gz, 0);
#endif
}

// ---------------------------------------------------------------------------
// fp32 -> bf16 conversion (4 elements / thread)
// ---------------------------------------------------------------------------
__global__ __launch_bounds__(256)
void cvt_f32_bf16_kernel(const float* __restrict__ src, bf16_t* __restrict__ dst, int n4) {
  int i = blockIdx.x * blockDim.x + threadIdx.x;
  if (i < n4) {
    v4f f = *(const v4f*)(src + (size_t)i * 4);
    *(v4bf*)(dst + (size_t)i * 4) = __builtin_convertvector(f, v4bf);
  }
}

// ---------------------------------------------------------------------------
// Fused Q/K/V projection: out = x @ W.T + b  (one wave per 64x64 tile)
// A (x rows) and B (W rows) tiles are TDM-staged into LDS, double-buffered.
// qkv layout: [mat][b][h][s][d] bf16; Q pre-scaled by 1/sqrt(HEAD_DIM)
// ---------------------------------------------------------------------------
__global__ __launch_bounds__(32)
void qkv_gemm_kernel(const bf16_t* __restrict__ xb,    // [M_TOTAL][D_MODEL]
                     const bf16_t* __restrict__ Wb,    // [3][D_MODEL][D_MODEL]
                     const float* __restrict__ bq,
                     const float* __restrict__ bk,
                     const float* __restrict__ bv,
                     bf16_t* __restrict__ qkv) {
  __shared__ __align__(16) bf16_t ash[2][64 * 32];     // A tiles (m-local x k-local)
  __shared__ __align__(16) bf16_t bsh[2][64 * 32];     // B tiles (n-local x k-local)

  const int lane = threadIdx.x;
  const int half = lane >> 4;
  const int ln16 = lane & 15;
  const int n0   = blockIdx.x * 64;
  const int m0   = blockIdx.y * 64;
  const int mat  = blockIdx.z;

  const bf16_t* W    = Wb + (size_t)mat * D_MODEL * D_MODEL;
  const float*  bias = (mat == 0) ? bq : (mat == 1 ? bk : bv);
  const float   scale = (mat == 0) ? 0.125f : 1.0f;    // 1/sqrt(64) folded into Q

  const bf16_t* Asrc = xb + (size_t)m0 * D_MODEL;
  const bf16_t* Bsrc = W + (size_t)n0 * D_MODEL;
  const u32 aoff[2] = {LDS_OFF(&ash[0][0]), LDS_OFF(&ash[1][0])};
  const u32 boff[2] = {LDS_OFF(&bsh[0][0]), LDS_OFF(&bsh[1][0])};

  // prologue: two A/B tile pairs in flight
  tdm_load_tile_64x32(Asrc + 0,  aoff[0]);
  tdm_load_tile_64x32(Bsrc + 0,  boff[0]);
  tdm_load_tile_64x32(Asrc + 32, aoff[1]);
  tdm_load_tile_64x32(Bsrc + 32, boff[1]);

  v8f acc[4][4];
#pragma unroll
  for (int ni = 0; ni < 4; ++ni) {
    float bv0 = bias[n0 + ni * 16 + ln16];
#pragma unroll
    for (int mi = 0; mi < 4; ++mi)
#pragma unroll
      for (int r = 0; r < 8; ++r) acc[mi][ni][r] = bv0;
  }

  for (int it = 0; it < D_MODEL / 32; ++it) {
    __builtin_amdgcn_s_wait_tensorcnt(2);   // oldest A/B pair landed (in-order per wave)
    const int buf = it & 1;

    v16bf afrag[4], bfrag[4];
#pragma unroll
    for (int mi = 0; mi < 4; ++mi) {
      const bf16_t* arow = &ash[buf][(mi * 16 + ln16) * 32];
      afrag[mi] = cat8(*(const v8bf*)(arow + half * 16),
                       *(const v8bf*)(arow + half * 16 + 8));
    }
#pragma unroll
    for (int ni = 0; ni < 4; ++ni)
      bfrag[ni] = *(const v16bf*)(&bsh[buf][(ni * 16 + ln16) * 32 + half * 16]);

#pragma unroll
    for (int mi = 0; mi < 4; ++mi)
#pragma unroll
      for (int ni = 0; ni < 4; ++ni)
        acc[mi][ni] = __builtin_amdgcn_wmma_f32_16x16x32_bf16(
            false, afrag[mi], false, bfrag[ni], (short)0, acc[mi][ni], false, false);

    // buffer reads done before TDM may overwrite it
    asm volatile("s_wait_dscnt 0" ::: "memory");
    const int kn = ((it + 2) & (D_MODEL / 32 - 1)) * 32;  // wrapped refill keeps pipe uniform
    tdm_load_tile_64x32(Asrc + kn, aoff[buf]);
    tdm_load_tile_64x32(Bsrc + kn, boff[buf]);
  }

#pragma unroll
  for (int mi = 0; mi < 4; ++mi) {
#pragma unroll
    for (int ni = 0; ni < 4; ++ni) {
      int n = n0 + ni * 16 + ln16;
      int h = n >> 6, d = n & 63;
#pragma unroll
      for (int r = 0; r < 8; ++r) {
        int m = m0 + mi * 16 + half * 8 + r;            // C/D layout: M = r + 8*(lane/16)
        int b = m >> 12;                                // / SEQ
        int s = m & (SEQ - 1);
        size_t off = (((size_t)(mat * BATCH + b) * NUM_HEADS + h) * SEQ + s) * HEAD_DIM + d;
        qkv[off] = (bf16_t)(acc[mi][ni][r] * scale);
      }
    }
  }
}

// ---------------------------------------------------------------------------
// Banded attention: one wave per (b, h, 16-query tile)
// ---------------------------------------------------------------------------
__global__ __launch_bounds__(32)
void attn_kernel(const bf16_t* __restrict__ qkv,      // [3][B][H][S][64]
                 bf16_t* __restrict__ ctx) {          // [B][S][D_MODEL]
  __shared__ __align__(16) float  sSh[16][KT_SPAN * 16 + 8]; // fp32 scores (+pad)
  __shared__ __align__(16) bf16_t pSh[16][PCOLS];            // bf16 probs, zero-padded

  const int lane = threadIdx.x;
  const int half = lane >> 4;
  const int ln16 = lane & 15;
  const int qt   = blockIdx.x;                // 0..255
  const int h    = blockIdx.y;
  const int b    = blockIdx.z;
  const int s0   = qt * 16;

  const size_t headOff = ((size_t)b * NUM_HEADS + h) * SEQ * HEAD_DIM;
  const size_t matStr  = (size_t)BATCH * NUM_HEADS * SEQ * HEAD_DIM;
  const bf16_t* Q = qkv + headOff;
  const bf16_t* K = qkv + matStr + headOff;
  const bf16_t* V = qkv + 2 * matStr + headOff;

  // Q fragments for this 16-row tile (K dim 64 -> two 32-chunks)
  const bf16_t* qrow = Q + (size_t)(s0 + ln16) * HEAD_DIM;
  v16bf qa0 = load_a_frag(qrow, 0, half);
  v16bf qa1 = load_a_frag(qrow, 32, half);

  // ---- scores: 9 key tiles, 2 WMMAs each, masked into LDS --------------------
  for (int t = 0; t < KT_SPAN; ++t) {
    int kt = qt - 4 + t;
    if (kt >= 0 && kt < SEQ / 16) {
      const bf16_t* krow = K + (size_t)(kt * 16 + ln16) * HEAD_DIM; // B col n = key
      v16bf kb0 = *(const v16bf*)(krow + half * 16);
      v16bf kb1 = *(const v16bf*)(krow + 32 + half * 16);
      v8f sc = {};
      sc = __builtin_amdgcn_wmma_f32_16x16x32_bf16(false, qa0, false, kb0, (short)0, sc, false, false);
      sc = __builtin_amdgcn_wmma_f32_16x16x32_bf16(false, qa1, false, kb1, (short)0, sc, false, false);
      int j = kt * 16 + ln16;
#pragma unroll
      for (int r = 0; r < 8; ++r) {
        int m  = half * 8 + r;
        int dq = (s0 + m) - j;
        if (dq < 0) dq = -dq;
        sSh[m][t * 16 + ln16] = (dq <= WINDOW) ? sc[r] : -__builtin_inff();
      }
    } else {
#pragma unroll
      for (int r = 0; r < 8; ++r) sSh[half * 8 + r][t * 16 + ln16] = -__builtin_inff();
    }
  }
  __syncthreads();

  // ---- softmax (fp32): one row per lane<16 ----------------------------------
  if (lane < 16) {
    float mx = -__builtin_inff();
    for (int c = 0; c < KT_SPAN * 16; ++c) mx = fmaxf(mx, sSh[lane][c]);
    float sum = 0.0f;
    for (int c = 0; c < KT_SPAN * 16; ++c) {
      float e = __expf(sSh[lane][c] - mx);   // exp(-inf) = 0 for masked keys
      sSh[lane][c] = e;
      sum += e;
    }
    float inv = 1.0f / sum;                  // diagonal is always in-band -> sum > 0
    for (int c = 0; c < KT_SPAN * 16; ++c) pSh[lane][c] = (bf16_t)(sSh[lane][c] * inv);
    for (int c = KT_SPAN * 16; c < PCOLS; ++c) pSh[lane][c] = (bf16_t)0.0f;
  }
  __syncthreads();

  // ---- context = P (16x160) @ V (160x64), rows clamped (probs are 0 there) ---
  const int kbase0 = (qt - 4) * 16;          // may be negative at edges
  v8f out[4];
#pragma unroll
  for (int ni = 0; ni < 4; ++ni)
#pragma unroll
    for (int r = 0; r < 8; ++r) out[ni][r] = 0.0f;

  for (int c = 0; c < 5; ++c) {              // 5 K-chunks of 32
    const bf16_t* prow = &pSh[ln16][0];
    v16bf pa = cat8(*(const v8bf*)(prow + c * 32 + half * 16),
                    *(const v8bf*)(prow + c * 32 + half * 16 + 8));
#pragma unroll
    for (int ni = 0; ni < 4; ++ni) {
      int d = ni * 16 + ln16;                // B col n = head dim
      v16bf vb;
#pragma unroll
      for (int jj = 0; jj < 16; ++jj) {
        int kr = kbase0 + c * 32 + half * 16 + jj;
        kr = kr < 0 ? 0 : (kr >= SEQ ? SEQ - 1 : kr);
        vb[jj] = V[(size_t)kr * HEAD_DIM + d];
      }
      out[ni] = __builtin_amdgcn_wmma_f32_16x16x32_bf16(
          false, pa, false, vb, (short)0, out[ni], false, false);
    }
  }

#pragma unroll
  for (int ni = 0; ni < 4; ++ni) {
    int d = ni * 16 + ln16;
#pragma unroll
    for (int r = 0; r < 8; ++r) {
      int m = half * 8 + r;
      ctx[((size_t)b * SEQ + s0 + m) * D_MODEL + h * HEAD_DIM + d] = (bf16_t)out[ni][r];
    }
  }
}

// ---------------------------------------------------------------------------
// Output projection: d_out = ctx @ Wo.T + bo (fp32 result), TDM-staged tiles
// ---------------------------------------------------------------------------
__global__ __launch_bounds__(32)
void out_gemm_kernel(const bf16_t* __restrict__ cb,   // [M_TOTAL][D_MODEL]
                     const bf16_t* __restrict__ Wo,   // [D_MODEL][D_MODEL]
                     const float* __restrict__ bo,
                     float* __restrict__ out) {
  __shared__ __align__(16) bf16_t ash[2][64 * 32];
  __shared__ __align__(16) bf16_t bsh[2][64 * 32];

  const int lane = threadIdx.x;
  const int half = lane >> 4;
  const int ln16 = lane & 15;
  const int n0   = blockIdx.x * 64;
  const int m0   = blockIdx.y * 64;

  const bf16_t* Asrc = cb + (size_t)m0 * D_MODEL;
  const bf16_t* Bsrc = Wo + (size_t)n0 * D_MODEL;
  const u32 aoff[2] = {LDS_OFF(&ash[0][0]), LDS_OFF(&ash[1][0])};
  const u32 boff[2] = {LDS_OFF(&bsh[0][0]), LDS_OFF(&bsh[1][0])};

  tdm_load_tile_64x32(Asrc + 0,  aoff[0]);
  tdm_load_tile_64x32(Bsrc + 0,  boff[0]);
  tdm_load_tile_64x32(Asrc + 32, aoff[1]);
  tdm_load_tile_64x32(Bsrc + 32, boff[1]);

  v8f acc[4][4];
#pragma unroll
  for (int ni = 0; ni < 4; ++ni) {
    float bv0 = bo[n0 + ni * 16 + ln16];
#pragma unroll
    for (int mi = 0; mi < 4; ++mi)
#pragma unroll
      for (int r = 0; r < 8; ++r) acc[mi][ni][r] = bv0;
  }

  for (int it = 0; it < D_MODEL / 32; ++it) {
    __builtin_amdgcn_s_wait_tensorcnt(2);
    const int buf = it & 1;

    v16bf afrag[4], bfrag[4];
#pragma unroll
    for (int mi = 0; mi < 4; ++mi) {
      const bf16_t* arow = &ash[buf][(mi * 16 + ln16) * 32];
      afrag[mi] = cat8(*(const v8bf*)(arow + half * 16),
                       *(const v8bf*)(arow + half * 16 + 8));
    }
#pragma unroll
    for (int ni = 0; ni < 4; ++ni)
      bfrag[ni] = *(const v16bf*)(&bsh[buf][(ni * 16 + ln16) * 32 + half * 16]);

#pragma unroll
    for (int mi = 0; mi < 4; ++mi)
#pragma unroll
      for (int ni = 0; ni < 4; ++ni)
        acc[mi][ni] = __builtin_amdgcn_wmma_f32_16x16x32_bf16(
            false, afrag[mi], false, bfrag[ni], (short)0, acc[mi][ni], false, false);

    asm volatile("s_wait_dscnt 0" ::: "memory");
    const int kn = ((it + 2) & (D_MODEL / 32 - 1)) * 32;
    tdm_load_tile_64x32(Asrc + kn, aoff[buf]);
    tdm_load_tile_64x32(Bsrc + kn, boff[buf]);
  }

#pragma unroll
  for (int mi = 0; mi < 4; ++mi)
#pragma unroll
    for (int ni = 0; ni < 4; ++ni) {
      int n = n0 + ni * 16 + ln16;
#pragma unroll
      for (int r = 0; r < 8; ++r) {
        int m = m0 + mi * 16 + half * 8 + r;
        out[(size_t)m * D_MODEL + n] = acc[mi][ni][r];
      }
    }
}

// ---------------------------------------------------------------------------
extern "C" void kernel_launch(void* const* d_in, const int* in_sizes, int n_in,
                              void* d_out, int out_size, void* d_ws, size_t ws_size,
                              hipStream_t stream) {
  const float* x  = (const float*)d_in[0];
  const float* Wq = (const float*)d_in[1];
  const float* bq = (const float*)d_in[2];
  const float* Wk = (const float*)d_in[3];
  const float* bk = (const float*)d_in[4];
  const float* Wv = (const float*)d_in[5];
  const float* bv = (const float*)d_in[6];
  const float* Wo = (const float*)d_in[7];
  const float* bo = (const float*)d_in[8];
  float* out = (float*)d_out;

  // workspace carve-up (bf16 intermediates, ~44 MB total)
  char* p = (char*)d_ws;
  bf16_t* xb   = (bf16_t*)p; p += (size_t)M_TOTAL * D_MODEL * sizeof(bf16_t);
  bf16_t* Wbuf = (bf16_t*)p; p += (size_t)4 * D_MODEL * D_MODEL * sizeof(bf16_t); // q,k,v,o
  bf16_t* qkvb = (bf16_t*)p; p += (size_t)3 * M_TOTAL * D_MODEL * sizeof(bf16_t);
  bf16_t* ctx  = (bf16_t*)p;

  const size_t WSZ = (size_t)D_MODEL * D_MODEL;
  int nx4 = M_TOTAL * D_MODEL / 4;
  int nw4 = (int)(WSZ / 4);
  cvt_f32_bf16_kernel<<<(nx4 + 255) / 256, 256, 0, stream>>>(x, xb, nx4);
  cvt_f32_bf16_kernel<<<(nw4 + 255) / 256, 256, 0, stream>>>(Wq, Wbuf + 0 * WSZ, nw4);
  cvt_f32_bf16_kernel<<<(nw4 + 255) / 256, 256, 0, stream>>>(Wk, Wbuf + 1 * WSZ, nw4);
  cvt_f32_bf16_kernel<<<(nw4 + 255) / 256, 256, 0, stream>>>(Wv, Wbuf + 2 * WSZ, nw4);
  cvt_f32_bf16_kernel<<<(nw4 + 255) / 256, 256, 0, stream>>>(Wo, Wbuf + 3 * WSZ, nw4);

  qkv_gemm_kernel<<<dim3(D_MODEL / 64, M_TOTAL / 64, 3), 32, 0, stream>>>(
      xb, Wbuf, bq, bk, bv, qkvb);
  attn_kernel<<<dim3(SEQ / 16, NUM_HEADS, BATCH), 32, 0, stream>>>(qkvb, ctx);
  out_gemm_kernel<<<dim3(D_MODEL / 64, M_TOTAL / 64), 32, 0, stream>>>(
      ctx, Wbuf + 3 * WSZ, bo, out);
}
